// Mlp_83554293776511
// MI455X (gfx1250) — 8-bit matrix operand order repaired and checked by reading, compiled, not run on hardware
//
#include <hip/hip_runtime.h>


typedef __attribute__((ext_vector_type(8))) int v8i;
typedef int v4i32 __attribute__((vector_size(16)));
typedef __attribute__((address_space(1))) v4i32* gptr4;
typedef __attribute__((address_space(3))) v4i32* lptr4;

#define QP        127.0f
#define MIN_RANGE 1e-6f

#if defined(__gfx1250__) && __has_builtin(__builtin_amdgcn_global_load_async_to_lds_b128)
#define HAVE_ASYNC_LDS 1
#else
#define HAVE_ASYNC_LDS 0
#endif

__device__ __forceinline__ void cp16(char* l, const char* g) {
#if HAVE_ASYNC_LDS
    __builtin_amdgcn_global_load_async_to_lds_b128((gptr4)g, (lptr4)l, 0, 0);
#else
    *(int4*)l = *(const int4*)g;
#endif
}

__device__ __forceinline__ void async_wait() {
#if HAVE_ASYNC_LDS
#if __has_builtin(__builtin_amdgcn_s_wait_asynccnt)
    __builtin_amdgcn_s_wait_asynccnt(0);
#else
    asm volatile("s_wait_asynccnt 0x0" ::: "memory");
#endif
#endif
}

__device__ __forceinline__ float wave_max32(float m) {
#pragma unroll
    for (int off = 16; off > 0; off >>= 1)
        m = fmaxf(m, __shfl_xor(m, off, 32));
    return m;
}

__global__ void init_maxes_kernel(float* p) {
    if (threadIdx.x < 4) ((int*)p)[threadIdx.x] = 0;
}

__global__ void absmax_kernel(const float* __restrict__ x, size_t n4,
                              float* __restrict__ out) {
    const float4* x4 = (const float4*)x;
    size_t stride = (size_t)gridDim.x * blockDim.x;
    float m = 0.f;
    for (size_t i = (size_t)blockIdx.x * blockDim.x + threadIdx.x; i < n4; i += stride) {
        float4 v = x4[i];
        m = fmaxf(m, fmaxf(fmaxf(fabsf(v.x), fabsf(v.y)),
                           fmaxf(fabsf(v.z), fabsf(v.w))));
    }
    m = wave_max32(m);
    __shared__ float sm[8];
    const int lane = threadIdx.x & 31, wid = threadIdx.x >> 5;
    if (lane == 0) sm[wid] = m;
    __syncthreads();
    if (threadIdx.x == 0) {
        float b = sm[0];
        for (int i = 1; i < (int)(blockDim.x >> 5); ++i) b = fmaxf(b, sm[i]);
        atomicMax((int*)out, __float_as_int(b));
    }
}

__global__ void quant_kernel(const float* __restrict__ x, size_t n4,
                             const float* __restrict__ amax,
                             char* __restrict__ q) {
    const float scale = fmaxf(*amax, MIN_RANGE) / QP;
    const float4* x4 = (const float4*)x;
    int* q4 = (int*)q;
    size_t stride = (size_t)gridDim.x * blockDim.x;
    for (size_t i = (size_t)blockIdx.x * blockDim.x + threadIdx.x; i < n4; i += stride) {
        float4 v = x4[i];
        int a = (int)rintf(v.x / scale) & 0xff;
        int b = (int)rintf(v.y / scale) & 0xff;
        int c = (int)rintf(v.z / scale) & 0xff;
        int d = (int)rintf(v.w / scale) & 0xff;
        q4[i] = a | (b << 8) | (c << 16) | (d << 24);
    }
}

#define LDS_RS   80
#define LDS_BOFF 10240
#define LDS_BUF  20480

__device__ __forceinline__ v8i lds_a_frag(const char* sa, int m0, int lane) {
    const char* p = sa + (m0 + (lane & 15)) * LDS_RS + ((lane >> 4) << 4);
    int4 c0 = *(const int4*)(p);
    int4 c1 = *(const int4*)(p + 32);
    v8i f = {c0.x, c0.y, c0.z, c0.w, c1.x, c1.y, c1.z, c1.w};
    return f;
}

__device__ __forceinline__ v8i lds_b_frag(const char* sb, int n0, int lane) {
    const char* p = sb + (n0 + (lane & 15)) * LDS_RS + ((lane >> 4) << 4);
    int4 c0 = *(const int4*)(p);
    int4 c1 = *(const int4*)(p + 32);
    v8i f = {c0.x, c0.y, c0.z, c0.w, c1.x, c1.y, c1.z, c1.w};
    return f;
}

__device__ __forceinline__ void stage(char* sm,
                                      const char* __restrict__ Aq,
                                      const char* __restrict__ Wq,
                                      int aRow0, int nRow0, int K, int k0,
                                      int tid) {
#pragma unroll
    for (int inst = 0; inst < 2; ++inst) {
        const int chunk = inst * 256 + tid;
        const int row = chunk >> 2, c = (chunk & 3) << 4;
        cp16(sm + row * LDS_RS + c,
             Aq + (size_t)(aRow0 + row) * K + (k0 + c));
    }
#pragma unroll
    for (int inst = 0; inst < 2; ++inst) {
        const int chunk = inst * 256 + tid;
        const int row = chunk >> 2, c = (chunk & 3) << 4;
        cp16(sm + LDS_BOFF + row * LDS_RS + c,
             Wq + (size_t)(nRow0 + row) * K + (k0 + c));
    }
}

template <bool FUSE_GELU>
__global__ void __launch_bounds__(256, 1)
gemm_i8_kernel(const char* __restrict__ Aq,
               const char* __restrict__ Wq,
               const float* __restrict__ bias,
               const float* __restrict__ amaxA,
               const float* __restrict__ amaxW,
               float* __restrict__ Out,
               float* __restrict__ amaxOut,
               int M, int N, int K) {
    __shared__ char smem[2][LDS_BUF];

    const int tid  = threadIdx.x;
    const int lane = tid & 31;
    const int wid  = tid >> 5;
    const int aRow0 = blockIdx.x * 128;
    const int nRow0 = blockIdx.y * 128;
    const int wm0  = (wid & 3) * 32;
    const int wn0  = (wid >> 2) * 64;

    const float sA = fmaxf(*amaxA, MIN_RANGE) / QP;
    const float sW = fmaxf(*amaxW, MIN_RANGE) / QP;
    const float s  = sA * sW;

    v8i acc[2][4];
#pragma unroll
    for (int i = 0; i < 2; ++i)
#pragma unroll
        for (int j = 0; j < 4; ++j) acc[i][j] = (v8i)0;

    stage(smem[0], Aq, Wq, aRow0, nRow0, K, 0, tid);
    async_wait();
    __syncthreads();

    for (int k0 = 0; k0 < K; k0 += 64) {
        const int buf = (k0 >> 6) & 1;
        if (k0 + 64 < K)
            stage(smem[buf ^ 1], Aq, Wq, aRow0, nRow0, K, k0 + 64, tid);

        const char* sa = smem[buf];
        const char* sb = smem[buf] + LDS_BOFF;
        v8i a[2], b[4];
#pragma unroll
        for (int i = 0; i < 2; ++i)
            a[i] = lds_a_frag(sa, wm0 + 16 * i, lane);
#pragma unroll
        for (int j = 0; j < 4; ++j)
            b[j] = lds_b_frag(sb, wn0 + 16 * j, lane);
#pragma unroll
        for (int i = 0; i < 2; ++i)
#pragma unroll
            for (int j = 0; j < 4; ++j)
                acc[i][j] = __builtin_amdgcn_wmma_i32_16x16x64_iu8(
                     true, a[i],  true, b[j],
                    acc[i][j],  false,  false);

        async_wait();
        __syncthreads();
    }

    const int tm0 = aRow0 + wm0;
    const int tn0 = nRow0 + wn0;
    const int rowoff = (lane >> 4) << 3;
    const int coloff = lane & 15;
    float lmax = 0.f;
#pragma unroll
    for (int j = 0; j < 4; ++j) {
        const int n = tn0 + 16 * j + coloff;
        const float bn = bias[n];
#pragma unroll
        for (int i = 0; i < 2; ++i) {
            const int mbase = tm0 + 16 * i + rowoff;
#pragma unroll
            for (int r = 0; r < 8; ++r) {
                float v = ((float)acc[i][j][r] + bn) * s;
                if (FUSE_GELU) {
                    v = 0.5f * v * (1.0f + erff(v * 0.7071067811865475f));
                    lmax = fmaxf(lmax, fabsf(v));
                }
                Out[(size_t)(mbase + r) * N + n] = v;
            }
        }
    }
    if (FUSE_GELU) {
        lmax = wave_max32(lmax);
        if (lane == 0) atomicMax((int*)amaxOut, __float_as_int(lmax));
    }
}

extern "C" void kernel_launch(void* const* d_in, const int* in_sizes, int n_in,
                              void* d_out, int out_size, void* d_ws, size_t ws_size,
                              hipStream_t stream) {
    const float* x  = (const float*)d_in[0];
    const float* w1 = (const float*)d_in[1];
    const float* b1 = (const float*)d_in[2];
    const float* w2 = (const float*)d_in[3];
    const float* b2 = (const float*)d_in[4];

    const int M = 8192;
    const int D = 1024;
    const int H = 4096;

    char*  ws   = (char*)d_ws;
    float* amax = (float*)ws;
    size_t off  = 256;
    char*  xq   = ws + off; off += (size_t)M * D;
    char*  w1q  = ws + off; off += (size_t)H * D;
    char*  w2q  = ws + off; off += (size_t)D * H;
    float* h    = (float*)(ws + off); off += (size_t)M * H * 4;
    char*  hq   = ws + off; off += (size_t)M * H;
    (void)ws_size; (void)in_sizes; (void)n_in; (void)out_size;

    init_maxes_kernel<<<1, 64, 0, stream>>>(amax);

    const int T = 256;
    absmax_kernel<<<1024, T, 0, stream>>>(x,  (size_t)M * D / 4, amax + 0);
    absmax_kernel<<<512,  T, 0, stream>>>(w1, (size_t)H * D / 4, amax + 1);
    absmax_kernel<<<512,  T, 0, stream>>>(w2, (size_t)D * H / 4, amax + 3);

    quant_kernel<<<(int)(((size_t)M * D / 4 + T - 1) / T), T, 0, stream>>>(
        x, (size_t)M * D / 4, amax + 0, xq);
    quant_kernel<<<(int)(((size_t)H * D / 4 + T - 1) / T), T, 0, stream>>>(
        w1, (size_t)H * D / 4, amax + 1, w1q);
    quant_kernel<<<(int)(((size_t)D * H / 4 + T - 1) / T), T, 0, stream>>>(
        w2, (size_t)D * H / 4, amax + 3, w2q);

    gemm_i8_kernel<true><<<dim3(M / 128, H / 128), 256, 0, stream>>>(
        xq, w1q, b1, amax + 0, amax + 1, h, amax + 2, M, H, D);

    quant_kernel<<<(int)(((size_t)M * H / 4 + T - 1) / T), T, 0, stream>>>(
        h, (size_t)M * H / 4, amax + 2, hq);

    gemm_i8_kernel<false><<<dim3(M / 128, D / 128), 256, 0, stream>>>(
        hq, w2q, b2, amax + 2, amax + 3, (float*)d_out, nullptr, M, D, H);
}
